// CrossAttention_50251117363418
// MI455X (gfx1250) — compile-verified, matrix operand order checked by reading, not run on hardware
//
#include <hip/hip_runtime.h>
#include <hip/hip_bf16.h>
#include <stdint.h>

typedef unsigned short u16;
typedef __attribute__((ext_vector_type(16))) __bf16 v16bf;
typedef __attribute__((ext_vector_type(8)))  float  v8f;
typedef __attribute__((ext_vector_type(4)))  unsigned int v4u;
typedef __attribute__((ext_vector_type(8)))  int v8i;
typedef __attribute__((ext_vector_type(4)))  int v4i;

#if defined(__gfx1250__) && __has_builtin(__builtin_amdgcn_tensor_load_to_lds) && \
    __has_builtin(__builtin_amdgcn_s_wait_tensorcnt)
#define USE_TDM 1
#else
#define USE_TDM 0
#endif

struct alignas(16) U4 { unsigned a[4]; };
struct alignas(16) F4 { float a[4]; };

union AFrag { v16bf v; U4 q[2]; };

__device__ __forceinline__ u16 f2bf(float f) {
  unsigned u = __builtin_bit_cast(unsigned, f);
  unsigned r = u + 0x7FFFu + ((u >> 16) & 1u);   // round-to-nearest-even
  return (u16)(r >> 16);
}
__device__ __forceinline__ float bf2f(u16 h) {
  return __builtin_bit_cast(float, (unsigned)h << 16);
}

// Stage 16 K-elements of one row into LDS (bf16), converting fp32 on the fly.
__device__ __forceinline__ void load16(const float* __restrict__ src, u16* dst) {
#pragma unroll
  for (int i = 0; i < 4; ++i) {
    F4 f = *(const F4*)(src + i * 4);
    unsigned p0 = (unsigned)f2bf(f.a[0]) | ((unsigned)f2bf(f.a[1]) << 16);
    unsigned p1 = (unsigned)f2bf(f.a[2]) | ((unsigned)f2bf(f.a[3]) << 16);
    *(unsigned*)(dst + i * 4)     = p0;
    *(unsigned*)(dst + i * 4 + 2) = p1;
  }
}
__device__ __forceinline__ void load16(const u16* __restrict__ src, u16* dst) {
  *(U4*)dst       = *(const U4*)src;
  *(U4*)(dst + 8) = *(const U4*)(src + 8);
}
__device__ __forceinline__ void storeOut(float* p, float v) { *p = v; }
__device__ __forceinline__ void storeOut(u16* p, float v)   { *p = f2bf(v); }

#if USE_TDM
// Issue one 2D TDM tile load (global -> LDS) via a hand-packed D#.
// tile dims in elements (data_size = 2 bytes), stride0 in elements.
// pad_int: DWORD-interval code (0=2,1=4,...,5=64); pad_amt: DWORDs-1.
__device__ __forceinline__ void tdm_load_2d(unsigned lds_off, const u16* gsrc,
                                            unsigned tile_d0, unsigned tile_d1,
                                            unsigned stride0,
                                            unsigned pad_int, unsigned pad_amt) {
  unsigned long long ga = (unsigned long long)(uintptr_t)gsrc;
  v4u g0;
  g0[0] = 1u;                                    // count=1, user descriptor
  g0[1] = lds_off;                               // LDS byte address
  g0[2] = (unsigned)(ga & 0xFFFFFFFFu);          // global_addr[31:0]
  g0[3] = ((unsigned)(ga >> 32) & 0x1FFFFFFu)    // global_addr[56:32]
          | (2u << 30);                          // type = 2 ("image")
  v8i g1;
  g1[0] = (int)((1u << 16)                       // data_size = 2 bytes
                | (1u << 20)                     // pad_enable
                | (pad_int << 22) | (pad_amt << 25));
  g1[1] = (int)((tile_d0 & 0xFFFFu) << 16);      // tensor_dim0[15:0] (== tile_d0)
  g1[2] = (int)(((tile_d0 >> 16) & 0xFFFFu)      // tensor_dim0[31:16]
                | ((tile_d1 & 0xFFFFu) << 16));  // tensor_dim1[15:0] (== tile_d1)
  g1[3] = (int)(((tile_d1 >> 16) & 0xFFFFu)      // tensor_dim1[31:16]
                | ((tile_d0 & 0xFFFFu) << 16));  // tile_dim0
  g1[4] = (int)(tile_d1 & 0xFFFFu);              // tile_dim1 (tile_dim2 = 0)
  g1[5] = (int)stride0;                          // tensor_dim0_stride[31:0]
  g1[6] = 0;
  g1[7] = 0;
  v4i gz4 = {0, 0, 0, 0};                        // groups 2/3 unused (2D)
  v8i gz8 = {0, 0, 0, 0, 0, 0, 0, 0};
  __builtin_amdgcn_tensor_load_to_lds(g0, g1, gz4, gz4, gz8, 0);
}
#endif

// ---------------------------------------------------------------------------
// GEMM: C[M,N] = A[M,K] * W[N,K]^T   (row-major, weights stored [out,in])
// bf16 WMMA, fp32 accumulate. 256 threads = 8 waves, 128x128 block tile,
// K-step 32. Wave grid 2x4: each wave owns 64x32 = 4x2 WMMA tiles.
// ---------------------------------------------------------------------------
#define LDA 40   // 32 + 8 pad (keeps 16B alignment: 80B row stride)

template <typename TA, typename TOut>
__global__ __launch_bounds__(256) void gemm_bf16_wmma(
    const TA* __restrict__ A, const float* __restrict__ W,
    TOut* __restrict__ C, int M, int N, int K) {
  __shared__ u16 As[128 * LDA];
  __shared__ u16 Bs[128 * LDA];

  const int tid  = threadIdx.x;
  const int wid  = tid >> 5;
  const int lane = tid & 31;
  const int lrow = lane & 15;
  const int grp  = lane >> 4;        // half-wave: K-base selector
  const int wr   = wid >> 2;         // 0..1  (M dir)
  const int wc   = wid & 3;          // 0..3  (N dir)
  const int mBase = blockIdx.y * 128;
  const int nBase = blockIdx.x * 128;
  const int ldr  = tid >> 1;         // loader row 0..127
  const int lseg = (tid & 1) * 16;   // loader K segment

  const v8f zacc = {0.f,0.f,0.f,0.f,0.f,0.f,0.f,0.f};
  v8f acc[4][2];
#pragma unroll
  for (int mi = 0; mi < 4; ++mi)
#pragma unroll
    for (int ni = 0; ni < 2; ++ni) acc[mi][ni] = zacc;

  for (int k0 = 0; k0 < K; k0 += 32) {
    __syncthreads();
    load16(A + (size_t)(mBase + ldr) * K + k0 + lseg, &As[ldr * LDA + lseg]);
    load16(W + (size_t)(nBase + ldr) * K + k0 + lseg, &Bs[ldr * LDA + lseg]);
    if (k0 + 32 < K) {
      __builtin_prefetch(A + (size_t)(mBase + ldr) * K + k0 + 32 + lseg, 0, 1);
      __builtin_prefetch(W + (size_t)(nBase + ldr) * K + k0 + 32 + lseg, 0, 1);
    }
    __syncthreads();

    AFrag af[4], bfr[2];
    const int kh = grp * 8;   // lanes 16-31 take K+8 (regs 0-3) / K+24 (regs 4-7)
#pragma unroll
    for (int mi = 0; mi < 4; ++mi) {
      const u16* p = &As[(wr * 64 + mi * 16 + lrow) * LDA + kh];
      af[mi].q[0] = *(const U4*)p;
      af[mi].q[1] = *(const U4*)(p + 16);
    }
#pragma unroll
    for (int ni = 0; ni < 2; ++ni) {
      const u16* p = &Bs[(wc * 32 + ni * 16 + lrow) * LDA + kh];
      bfr[ni].q[0] = *(const U4*)p;
      bfr[ni].q[1] = *(const U4*)(p + 16);
    }
#pragma unroll
    for (int mi = 0; mi < 4; ++mi)
#pragma unroll
      for (int ni = 0; ni < 2; ++ni)
        acc[mi][ni] = __builtin_amdgcn_wmma_f32_16x16x32_bf16(
            false, af[mi].v, false, bfr[ni].v, (short)0, acc[mi][ni], false, false);
  }

  // C layout: reg r, lane l -> row = r + (l>=16 ? 8 : 0), col = l&15
#pragma unroll
  for (int mi = 0; mi < 4; ++mi)
#pragma unroll
    for (int ni = 0; ni < 2; ++ni) {
      int row0 = mBase + wr * 64 + mi * 16 + grp * 8;
      int col  = nBase + wc * 32 + ni * 16 + lrow;
#pragma unroll
      for (int r = 0; r < 8; ++r)
        storeOut(&C[(size_t)(row0 + r) * N + col], acc[mi][ni][r]);
    }
}

// ---------------------------------------------------------------------------
// Per-head RMSNorm (128-wide rows). One wave per row; relayouts to
// [b][h][seq][128]. Fuses the attention scale (d^-0.5) for Q.
// ---------------------------------------------------------------------------
__global__ __launch_bounds__(256) void head_rmsnorm(
    const u16* __restrict__ in, const float* __restrict__ wvec,
    u16* __restrict__ out, int nheads, int seq, int incols, float scale,
    int totalRows) {
  int wid = threadIdx.x >> 5, lane = threadIdx.x & 31;
  int rowid = blockIdx.x * 8 + wid;
  if (rowid >= totalRows) return;
  int m = rowid / nheads, h = rowid % nheads;
  int b = m / seq, i = m % seq;
  const u16* src = in + (size_t)m * incols + h * 128;
  float x[4]; float ss = 0.f;
#pragma unroll
  for (int c = 0; c < 4; ++c) {
    x[c] = bf2f(src[c * 32 + lane]);
    ss += x[c] * x[c];
  }
#pragma unroll
  for (int off = 16; off >= 1; off >>= 1) ss += __shfl_xor(ss, off, 32);
  float rinv = rsqrtf(ss * (1.0f / 128.0f) + 1e-6f);
  u16* dst = out + (((size_t)b * nheads + h) * seq + i) * 128;
#pragma unroll
  for (int c = 0; c < 4; ++c) {
    int d = c * 32 + lane;
    dst[d] = f2bf(x[c] * rinv * wvec[d] * scale);
  }
}

// ---------------------------------------------------------------------------
// V repack: [b*1024, 8*128] -> [b][kvh][d][j]  (transposed per head so PV
// B-fragments read contiguous j-chunks from LDS)
// ---------------------------------------------------------------------------
__global__ __launch_bounds__(256) void v_repack(const u16* __restrict__ in,
                                                u16* __restrict__ out) {
  size_t idx = (size_t)blockIdx.x * 256 + threadIdx.x;  // 2*1024*1024 elems
  int m = (int)(idx >> 10);
  int c = (int)(idx & 1023);
  int b = m >> 10, j = m & 1023;
  int kvh = c >> 7, d = c & 127;
  out[(((size_t)b * 8 + kvh) * 128 + d) * 1024 + j] = in[idx];
}

// ---------------------------------------------------------------------------
// Fused flash attention. Block = 128 threads (4 waves), 64 query rows per
// block (16 per wave, kept as register A-fragments), KV tiles of 64 in LDS.
// K/V tiles are DMA'd by the Tensor Data Mover (TENSORcnt) when available.
// Online softmax in fp32; P relayouted C->A via per-wave LDS buffer.
// ---------------------------------------------------------------------------
__global__ __launch_bounds__(128) void attn_fused(
    const u16* __restrict__ qn, const u16* __restrict__ kn,
    const u16* __restrict__ vtr, const unsigned char* __restrict__ mask,
    u16* __restrict__ ao) {
  const int b = blockIdx.z, h = blockIdx.y;
  const int qb = blockIdx.x * 64;
  const int kvh = h / 3;  // GQA: 24 heads -> 8 kv heads, repeated consecutively
  const int tid = threadIdx.x, w = tid >> 5, lane = tid & 31;
  const int lrow = lane & 15, grp = lane >> 4;

  __shared__ u16 kt[64][136];     // [j][d]  (+4 DW pad per 64 DW row)
  __shared__ u16 vs[128][72];     // [d][j]  (+4 DW pad per 32 DW row)
  __shared__ u16 ps[4][16][72];   // per-wave P relayout buffer

  const v8f zacc = {0.f,0.f,0.f,0.f,0.f,0.f,0.f,0.f};

  // Q A-fragments for this wave's 16 rows, resident all kernel.
  AFrag qf[4];
  {
    const size_t qoff = (((size_t)b * 24 + h) * 2048 + qb + w * 16 + lrow) * 128;
#pragma unroll
    for (int kc = 0; kc < 4; ++kc) {
      const u16* p = qn + qoff + kc * 32 + grp * 8;
      qf[kc].q[0] = *(const U4*)p;
      qf[kc].q[1] = *(const U4*)(p + 16);
    }
  }
  bool rowok[8];
#pragma unroll
  for (int r = 0; r < 8; ++r)
    rowok[r] = mask[(size_t)b * 2048 + qb + w * 16 + grp * 8 + r] != 0;

  float mrun[8], lrun[8];
  v8f o[8];
#pragma unroll
  for (int r = 0; r < 8; ++r) { mrun[r] = -1e30f; lrun[r] = 0.f; }
#pragma unroll
  for (int di = 0; di < 8; ++di) o[di] = zacc;

  const size_t kbase = ((size_t)b * 8 + kvh) * 1024 * 128;
  const size_t vbase = ((size_t)b * 8 + kvh) * 128 * 1024;

  for (int jt = 0; jt < 1024; jt += 64) {
    __syncthreads();   // previous tile fully consumed
#if USE_TDM
    if (w == 0) {
      // K tile: 64 rows(j) x 128(d); LDS row = 64 DW data + 4 DW pad (136 u16)
      tdm_load_2d((unsigned)(uintptr_t)&kt[0][0],
                  kn + kbase + (size_t)jt * 128,
                  /*tile_d0=*/128, /*tile_d1=*/64, /*stride0=*/128,
                  /*pad_int=64DW*/5u, /*pad_amt=4DW*/3u);
      // V tile: 128 rows(d) x 64(j); LDS row = 32 DW data + 4 DW pad (72 u16)
      tdm_load_2d((unsigned)(uintptr_t)&vs[0][0],
                  vtr + vbase + jt,
                  /*tile_d0=*/64, /*tile_d1=*/128, /*stride0=*/1024,
                  /*pad_int=32DW*/4u, /*pad_amt=4DW*/3u);
      __builtin_amdgcn_s_wait_tensorcnt(0);
    }
    __syncthreads();   // publish TDM-written tiles to all waves
#else
    {  // K tile: 64 rows(j) x 128 cols(d)
      int row = tid & 63, half = tid >> 6;
      const u16* src = kn + kbase + (size_t)(jt + row) * 128 + half * 64;
      u16* dst = &kt[row][half * 64];
#pragma unroll
      for (int c = 0; c < 8; ++c) *(U4*)(dst + c * 8) = *(const U4*)(src + c * 8);
    }
    {  // V tile (pre-transposed source): 128 rows(d) x 64 cols(j)
      const u16* src = vtr + vbase + (size_t)tid * 1024 + jt;
#pragma unroll
      for (int c = 0; c < 8; ++c) *(U4*)(&vs[tid][c * 8]) = *(const U4*)(src + c * 8);
    }
    __syncthreads();
#endif

    // S = Q K^T : 4 j-subtiles x (K=128 in 4 chunks of 32)
    v8f s[4];
#pragma unroll
    for (int ni = 0; ni < 4; ++ni) {
      v8f accs = zacc;
#pragma unroll
      for (int kc = 0; kc < 4; ++kc) {
        AFrag bk;
        const u16* p = &kt[ni * 16 + lrow][kc * 32 + grp * 8];
        bk.q[0] = *(const U4*)p;
        bk.q[1] = *(const U4*)(p + 16);
        accs = __builtin_amdgcn_wmma_f32_16x16x32_bf16(
            false, qf[kc].v, false, bk.v, (short)0, accs, false, false);
      }
      s[ni] = accs;
    }

    // Row max (reg r + half-wave group -> row), 16-lane xor reduction.
    float tmax[8];
#pragma unroll
    for (int r = 0; r < 8; ++r) {
      float v = -1e30f;
#pragma unroll
      for (int ni = 0; ni < 4; ++ni) {
        float sv = rowok[r] ? s[ni][r] : 0.0f;  // masked row -> uniform softmax
        s[ni][r] = sv;
        v = fmaxf(v, sv);
      }
      v = fmaxf(v, __shfl_xor(v, 1, 32));
      v = fmaxf(v, __shfl_xor(v, 2, 32));
      v = fmaxf(v, __shfl_xor(v, 4, 32));
      v = fmaxf(v, __shfl_xor(v, 8, 32));
      tmax[r] = v;
    }
    float scl[8];
#pragma unroll
    for (int r = 0; r < 8; ++r) {
      float mn = fmaxf(mrun[r], tmax[r]);
      scl[r] = __expf(mrun[r] - mn);
      mrun[r] = mn;
    }
    float tsum[8];
#pragma unroll
    for (int r = 0; r < 8; ++r) tsum[r] = 0.f;
#pragma unroll
    for (int ni = 0; ni < 4; ++ni) {
#pragma unroll
      for (int r = 0; r < 8; ++r) {
        float p = __expf(s[ni][r] - mrun[r]);
        tsum[r] += p;
        ps[w][grp * 8 + r][ni * 16 + lrow] = f2bf(p);
      }
    }
#pragma unroll
    for (int r = 0; r < 8; ++r) {
      float t = tsum[r];
      t += __shfl_xor(t, 1, 32);
      t += __shfl_xor(t, 2, 32);
      t += __shfl_xor(t, 4, 32);
      t += __shfl_xor(t, 8, 32);
      lrun[r] = lrun[r] * scl[r] + t;
    }
#pragma unroll
    for (int di = 0; di < 8; ++di)
#pragma unroll
      for (int r = 0; r < 8; ++r) o[di][r] *= scl[r];

    // O += P V : 8 d-subtiles x 2 j-chunks of 32
#pragma unroll
    for (int di = 0; di < 8; ++di) {
#pragma unroll
      for (int jc = 0; jc < 2; ++jc) {
        AFrag pa, vb;
        const u16* pp = &ps[w][lrow][jc * 32 + grp * 8];
        pa.q[0] = *(const U4*)pp;
        pa.q[1] = *(const U4*)(pp + 16);
        const u16* vp = &vs[di * 16 + lrow][jc * 32 + grp * 8];
        vb.q[0] = *(const U4*)vp;
        vb.q[1] = *(const U4*)(vp + 16);
        o[di] = __builtin_amdgcn_wmma_f32_16x16x32_bf16(
            false, pa.v, false, vb.v, (short)0, o[di], false, false);
      }
    }
  }

  float inv[8];
#pragma unroll
  for (int r = 0; r < 8; ++r) inv[r] = 1.0f / lrun[r];
#pragma unroll
  for (int di = 0; di < 8; ++di)
#pragma unroll
    for (int r = 0; r < 8; ++r) {
      size_t row = (size_t)b * 2048 + qb + w * 16 + grp * 8 + r;
      ao[row * 3072 + h * 128 + di * 16 + lrow] = f2bf(o[di][r] * inv[r]);
    }
}

// ---------------------------------------------------------------------------
extern "C" void kernel_launch(void* const* d_in, const int* in_sizes, int n_in,
                              void* d_out, int out_size, void* d_ws, size_t ws_size,
                              hipStream_t stream) {
  const float* hidden = (const float*)d_in[0];            // [2,2048,3072]
  const float* vfeat  = (const float*)d_in[1];            // [2,1024,3072]
  const unsigned char* amask = (const unsigned char*)d_in[2]; // [2,2048] bool
  const float* Wq = (const float*)d_in[3];                // [3072,3072]
  const float* Wk = (const float*)d_in[4];                // [1024,3072]
  const float* Wv = (const float*)d_in[5];                // [1024,3072]
  const float* Wo = (const float*)d_in[6];                // [3072,3072]
  const float* qw = (const float*)d_in[7];                // [128]
  const float* kw = (const float*)d_in[8];                // [128]
  float* out = (float*)d_out;                             // [2,2048,3072]

  char* ws = (char*)d_ws;
  u16* qb16 = (u16*)(ws + 0);          // 4096x3072 bf16  (25,165,824 B)
  u16* kb16 = (u16*)(ws + 25165824);   // 2048x1024 bf16  ( 4,194,304 B)
  u16* vb16 = (u16*)(ws + 29360128);   // 2048x1024 bf16
  u16* qnrm = (u16*)(ws + 33554432);   // [b][24][2048][128] bf16
  u16* knrm = (u16*)(ws + 58720256);   // [b][8][1024][128]  bf16
  u16* vtr  = (u16*)(ws + 62914560);   // [b][8][128][1024]  bf16
  u16* ao   = (u16*)(ws + 67108864);   // 4096x3072 bf16

  // Projections (fp32 -> bf16 convert in the GEMM loader)
  gemm_bf16_wmma<float, u16><<<dim3(24, 32), 256, 0, stream>>>(hidden, Wq, qb16, 4096, 3072, 3072);
  gemm_bf16_wmma<float, u16><<<dim3(8, 16),  256, 0, stream>>>(vfeat,  Wk, kb16, 2048, 1024, 3072);
  gemm_bf16_wmma<float, u16><<<dim3(8, 16),  256, 0, stream>>>(vfeat,  Wv, vb16, 2048, 1024, 3072);

  // Per-head RMSNorm (+ d^-0.5 fused into Q) and V transpose-repack
  head_rmsnorm<<<12288, 256, 0, stream>>>(qb16, qw, qnrm, 24, 2048, 3072,
                                          0.08838834764831845f, 98304);
  head_rmsnorm<<<2048, 256, 0, stream>>>(kb16, kw, knrm, 8, 1024, 1024, 1.0f, 16384);
  v_repack<<<8192, 256, 0, stream>>>(vb16, vtr);

  // Fused flash attention (TDM tile loads when available)
  attn_fused<<<dim3(32, 24, 2), 128, 0, stream>>>(qnrm, knrm, vtr, amask, ao);

  // Output projection (bf16 A, fp32 out)
  gemm_bf16_wmma<u16, float><<<dim3(24, 32), 256, 0, stream>>>(ao, Wo, out, 4096, 3072, 3072);
}
